// OmniAnomalyModel_40604620816465
// MI455X (gfx1250) — compile-verified, matrix operand order checked by reading, not run on hardware
//
#include <hip/hip_runtime.h>
#include <hip/hip_bf16.h>
#include <hip/hip_fp16.h>

// ---------------------------------------------------------------------------
// OmniAnomaly forward on MI455X (gfx1250, wave32).
// All K=500/503 GEMMs padded to K=512 and run through v_wmma_f32_16x16x32_f16
// (f16 inputs, f32 accumulate) with N-blocked register tiles (A-fragment
// reuse). Elementwise GRU/flow/head math stays f32.
// ---------------------------------------------------------------------------

typedef __attribute__((ext_vector_type(16))) _Float16 v16h;
typedef __attribute__((ext_vector_type(8)))  float    v8f;

constexpr int kB = 512, kT = 100, kC = 38, kH = 500, kD = 500, kZ = 3, kNF = 20;
constexpr int kHp  = 512;      // padded K for every WMMA GEMM
constexpr int kG3  = 1500, kG3p = 1504;
constexpr float kEPS = 1e-4f;
constexpr float kLOG2PI = 1.8378770664093453f;

// d_out layout (floats), reference tuple flattened in return order
constexpr size_t o_z0   = 0;
constexpr size_t o_z    = 153600;
constexpr size_t o_sld  = 307200;
constexpr size_t o_muq  = 358400;
constexpr size_t o_lvq  = 512000;
constexpr size_t o_mup  = 665600;
constexpr size_t o_lvp  = 819200;
constexpr size_t o_xmu  = 972800;
constexpr size_t o_xstd = 2918400;
constexpr size_t o_rlp  = 4864000;
constexpr size_t o_score= 4915200;

union FragU { uint4 q[2]; v16h v; };

// NB 16x16 f32 tiles sharing one A fragment per K-chunk.
// A row-major [*,512] f16; B = weights [N_pad,512] f16 (B-fragment element j
// = W[n][k_j], matching the 16-bit 32x16 B VGPR layout).
template <int NB>
__device__ inline void wmma_tiles(const _Float16* __restrict__ A,
                                  const _Float16* __restrict__ Bw,
                                  int mt, int nt0, v8f acc[NB]) {
  const int lane = threadIdx.x & 31;
  const int r = lane & 15, hi = lane >> 4;
  const _Float16* pa  = A  + (size_t)(mt * 16 + r) * kHp + hi * 8;
  const _Float16* pb0 = Bw + (size_t)(nt0 * 16 + r) * kHp + hi * 8;
#pragma unroll 2
  for (int k0 = 0; k0 < kHp; k0 += 32) {
    FragU ua;
    ua.q[0] = *(const uint4*)(pa);          // k = k0 + hi*8 .. +7
    ua.q[1] = *(const uint4*)(pa + 16);     // k = k0 + 16 + hi*8 .. +7
#pragma unroll
    for (int j = 0; j < NB; ++j) {
      const _Float16* pb = pb0 + (size_t)j * 16 * kHp;
      FragU ub;
      ub.q[0] = *(const uint4*)(pb);
      ub.q[1] = *(const uint4*)(pb + 16);
      acc[j] = __builtin_amdgcn_wmma_f32_16x16x32_f16(false, ua.v, false, ub.v,
                                                      (short)0, acc[j], false, false);
    }
    pa += 32; pb0 += 32;
  }
}

__device__ inline v8f vzero8() {
  v8f z = {0.f, 0.f, 0.f, 0.f, 0.f, 0.f, 0.f, 0.f};
  return z;
}

__device__ inline float sigm(float x) { return 1.f / (1.f + expf(-x)); }
__device__ inline float softplusf(float x) { return (x > 20.f) ? x : log1pf(expf(x)); }

// ---------------------------------------------------------------------------
__global__ void zero_k(uint4* __restrict__ p, int n) {
  int i = blockIdx.x * 256 + threadIdx.x;
  if (i < n) p[i] = make_uint4(0u, 0u, 0u, 0u);
}

// f32 [N,K] -> f16 [Np,Kp] zero padded
__global__ void pack_w_k(const float* __restrict__ W, _Float16* __restrict__ outp,
                         int N, int K, int Kp, int total) {
  int idx = blockIdx.x * 256 + threadIdx.x;
  if (idx >= total) return;
  int n = idx / Kp, k = idx % Kp;
  float v = (n < N && k < K) ? W[(size_t)n * K + k] : 0.f;
  outp[idx] = (_Float16)v;
}

// G[512,1504] = h_f16[512,512] @ Whh_h[1504,512]^T  (raw, bias added later)
// NB=2: 32 m-tiles x 47 n-groups = 1504 waves -> 188 blocks
__global__ void gemm_gates_k(const _Float16* __restrict__ A,
                             const _Float16* __restrict__ Bw,
                             float* __restrict__ G) {
  int wid = blockIdx.x * 8 + (threadIdx.x >> 5);
  int mt = wid / 47, ntb = wid % 47;
  v8f acc[2] = {vzero8(), vzero8()};
  wmma_tiles<2>(A, Bw, mt, ntb * 2, acc);
  int lane = threadIdx.x & 31;
  int m0 = mt * 16 + ((lane >> 4) << 3);
#pragma unroll
  for (int j = 0; j < 2; ++j) {
    int n = (ntb * 2 + j) * 16 + (lane & 15);
#pragma unroll
    for (int v = 0; v < 8; ++v)
      G[(size_t)(m0 + v) * kG3p + n] = acc[j][v];
  }
}

// qh[512,512] = relu(hseq_t @ qhW^T + z_prev @ qhW_z^T + qhb)  (z part in epilogue)
// NB=4: 32 m-tiles x 8 n-groups = 256 waves -> 32 blocks
__global__ void gemm_qh_k(const _Float16* __restrict__ A, const _Float16* __restrict__ Bw,
                          const float* __restrict__ qhW, const float* __restrict__ qhb,
                          const float* __restrict__ zprev, float* __restrict__ qh) {
  int wid = blockIdx.x * 8 + (threadIdx.x >> 5);
  int mt = wid >> 3, nt0 = (wid & 7) * 4;
  v8f acc[4] = {vzero8(), vzero8(), vzero8(), vzero8()};
  wmma_tiles<4>(A, Bw, mt, nt0, acc);
  int lane = threadIdx.x & 31;
  int m0 = mt * 16 + ((lane >> 4) << 3);
  float w0[4], w1[4], w2[4], bias[4];
  int nn[4];
#pragma unroll
  for (int j = 0; j < 4; ++j) {
    int n = (nt0 + j) * 16 + (lane & 15);
    nn[j] = n;
    w0[j] = w1[j] = w2[j] = bias[j] = 0.f;
    if (n < kD) {
      w0[j] = qhW[(size_t)n * 503 + 500];
      w1[j] = qhW[(size_t)n * 503 + 501];
      w2[j] = qhW[(size_t)n * 503 + 502];
      bias[j] = qhb[n];
    }
  }
#pragma unroll
  for (int v = 0; v < 8; ++v) {
    int m = m0 + v;
    float zp0 = zprev[m * 4 + 0], zp1 = zprev[m * 4 + 1], zp2 = zprev[m * 4 + 2];
#pragma unroll
    for (int j = 0; j < 4; ++j) {
      float val = 0.f;
      if (nn[j] < kD) {
        val = acc[j][v] + bias[j] + zp0 * w0[j] + zp1 * w1[j] + zp2 * w2[j];
        val = fmaxf(val, 0.f);
      }
      qh[(size_t)m * kHp + nn[j]] = val;
    }
  }
}

// dh_f16[51200,512] = relu(dec_seq @ dhW^T + dhb) (pad cols zeroed)
// NB=4: 3200 m-tiles x 8 n-groups = 25600 waves -> 3200 blocks
__global__ void gemm_dh_k(const _Float16* __restrict__ A, const _Float16* __restrict__ Bw,
                          const float* __restrict__ bias, _Float16* __restrict__ dh) {
  int wid = blockIdx.x * 8 + (threadIdx.x >> 5);
  int mt = wid >> 3, nt0 = (wid & 7) * 4;
  v8f acc[4] = {vzero8(), vzero8(), vzero8(), vzero8()};
  wmma_tiles<4>(A, Bw, mt, nt0, acc);
  int lane = threadIdx.x & 31;
  int m0 = mt * 16 + ((lane >> 4) << 3);
#pragma unroll
  for (int j = 0; j < 4; ++j) {
    int n = (nt0 + j) * 16 + (lane & 15);
    float bn = (n < kD) ? bias[n] : 0.f;
#pragma unroll
    for (int v = 0; v < 8; ++v) {
      float val = (n < kD) ? fmaxf(acc[j][v] + bn, 0.f) : 0.f;
      dh[(size_t)(m0 + v) * kHp + n] = (_Float16)val;
    }
  }
}

// x_mu / x_std head: rows m = t*512 + b, output stored [b][t][c]
// NB=3 covers all 48 output cols: 3200 waves -> 400 blocks
__global__ void gemm_head_k(const _Float16* __restrict__ A, const _Float16* __restrict__ Bw,
                            const float* __restrict__ bias, float* __restrict__ outp,
                            int mode) {
  int wid = blockIdx.x * 8 + (threadIdx.x >> 5);
  int mt = wid;
  v8f acc[3] = {vzero8(), vzero8(), vzero8()};
  wmma_tiles<3>(A, Bw, mt, 0, acc);
  int lane = threadIdx.x & 31;
  int m0 = mt * 16 + ((lane >> 4) << 3);
#pragma unroll
  for (int j = 0; j < 3; ++j) {
    int n = j * 16 + (lane & 15);
    if (n < kC) {
      float bn = bias[n];
#pragma unroll
      for (int v = 0; v < 8; ++v) {
        int m = m0 + v;
        int t = m >> 9, b = m & 511;
        float val = acc[j][v] + bn;
        if (mode) val = softplusf(val) + kEPS;
        outp[((size_t)b * kT + t) * kC + n] = val;
      }
    }
  }
}

// shared GRU tail
__device__ inline void gru_tail(int idx, int b, int i, float xr, float xz, float xn,
                                const float* __restrict__ G, const float* __restrict__ bhh,
                                const float* __restrict__ hprev, float* __restrict__ hnext,
                                _Float16* __restrict__ hnext_h, _Float16* __restrict__ hseq_t) {
  float hr = G[(size_t)b * kG3p + i]          + bhh[i];
  float hz = G[(size_t)b * kG3p + kH + i]     + bhh[kH + i];
  float hn = G[(size_t)b * kG3p + 2 * kH + i] + bhh[2 * kH + i];
  float rg = sigm(xr + hr);
  float ug = sigm(xz + hz);
  float ng = tanhf(xn + rg * hn);
  float h  = hprev[idx];
  float hw = (1.f - ug) * ng + ug * h;
  hnext[idx]   = hw;
  hnext_h[idx] = (_Float16)hw;
  hseq_t[idx]  = (_Float16)hw;
}

__global__ void gru_update_enc_k(const float* __restrict__ x, int t,
                                 const float* __restrict__ Wih, const float* __restrict__ bih,
                                 const float* __restrict__ bhh, const float* __restrict__ G,
                                 const float* __restrict__ hprev, float* __restrict__ hnext,
                                 _Float16* __restrict__ hnext_h, _Float16* __restrict__ hseq_t) {
  int idx = blockIdx.x * 256 + threadIdx.x;  // 512*512
  int b = idx >> 9, i = idx & 511;
  if (i >= kH) { hnext_h[idx] = (_Float16)0.f; hseq_t[idx] = (_Float16)0.f; return; }
  const float* xrow = x + ((size_t)b * kT + t) * kC;
  const float* wr = Wih + (size_t)i * kC;
  const float* wz = Wih + (size_t)(kH + i) * kC;
  const float* wn = Wih + (size_t)(2 * kH + i) * kC;
  float xr = bih[i], xz = bih[kH + i], xn = bih[2 * kH + i];
  for (int c = 0; c < kC; ++c) {
    float xv = xrow[c];
    xr = fmaf(xv, wr[c], xr);
    xz = fmaf(xv, wz[c], xz);
    xn = fmaf(xv, wn[c], xn);
  }
  gru_tail(idx, b, i, xr, xz, xn, G, bhh, hprev, hnext, hnext_h, hseq_t);
}

__global__ void gru_update_dec_k(const float* __restrict__ zall, int t,
                                 const float* __restrict__ Wih, const float* __restrict__ bih,
                                 const float* __restrict__ bhh, const float* __restrict__ G,
                                 const float* __restrict__ hprev, float* __restrict__ hnext,
                                 _Float16* __restrict__ hnext_h, _Float16* __restrict__ hseq_t) {
  int idx = blockIdx.x * 256 + threadIdx.x;
  int b = idx >> 9, i = idx & 511;
  if (i >= kH) { hnext_h[idx] = (_Float16)0.f; hseq_t[idx] = (_Float16)0.f; return; }
  const float* zb = zall + ((size_t)b * kT + t) * kZ;
  float z0v = zb[0], z1v = zb[1], z2v = zb[2];
  float xr = bih[i]        + z0v * Wih[i*3]        + z1v * Wih[i*3+1]        + z2v * Wih[i*3+2];
  float xz = bih[kH + i]   + z0v * Wih[(kH+i)*3]   + z1v * Wih[(kH+i)*3+1]   + z2v * Wih[(kH+i)*3+2];
  float xn = bih[2*kH + i] + z0v * Wih[(2*kH+i)*3] + z1v * Wih[(2*kH+i)*3+1] + z2v * Wih[(2*kH+i)*3+2];
  gru_tail(idx, b, i, xr, xz, xn, G, bhh, hprev, hnext, hnext_h, hseq_t);
}

// posterior heads + reparameterized sample, one wave per batch row
__global__ void post_head_k(const float* __restrict__ qh, const float* __restrict__ qmuW,
                            const float* __restrict__ qmub, const float* __restrict__ qlvW,
                            const float* __restrict__ qlvb, const float* __restrict__ noise,
                            int t, float* __restrict__ zprev_next, float* __restrict__ out) {
  int b = blockIdx.x * 8 + (threadIdx.x >> 5);
  int lane = threadIdx.x & 31;
  const float* row = qh + (size_t)b * kHp;
  float a0=0,a1=0,a2=0,a3=0,a4=0,a5=0;
  for (int d = lane; d < kD; d += 32) {
    float v = row[d];
    a0 = fmaf(v, qmuW[d],        a0);
    a1 = fmaf(v, qmuW[kD + d],   a1);
    a2 = fmaf(v, qmuW[2*kD + d], a2);
    a3 = fmaf(v, qlvW[d],        a3);
    a4 = fmaf(v, qlvW[kD + d],   a4);
    a5 = fmaf(v, qlvW[2*kD + d], a5);
  }
  for (int o = 16; o; o >>= 1) {
    a0 += __shfl_xor(a0, o, 32); a1 += __shfl_xor(a1, o, 32);
    a2 += __shfl_xor(a2, o, 32); a3 += __shfl_xor(a3, o, 32);
    a4 += __shfl_xor(a4, o, 32); a5 += __shfl_xor(a5, o, 32);
  }
  if (lane == 0) {
    float mu[3] = {a0 + qmub[0], a1 + qmub[1], a2 + qmub[2]};
    float lv[3] = {a3 + qlvb[0], a4 + qlvb[1], a5 + qlvb[2]};
#pragma unroll
    for (int j = 0; j < 3; ++j) {
      float lvj = fminf(fmaxf(lv[j], -8.f), 8.f);
      float sd  = fmaxf(expf(0.5f * lvj), kEPS);
      float e   = noise[((size_t)b * kT + t) * kZ + j];
      float z0v = mu[j] + e * sd;
      size_t o3 = ((size_t)b * kT + t) * kZ + j;
      out[o_z0  + o3] = z0v;
      out[o_muq + o3] = mu[j];
      out[o_lvq + o3] = lvj;
      zprev_next[b * 4 + j] = z0v;
    }
  }
}

// 20 planar flows, one thread per (b,t)
__global__ void flows_k(const float* __restrict__ fu, const float* __restrict__ fw,
                        const float* __restrict__ fb, float* __restrict__ out) {
  int idx = blockIdx.x * 256 + threadIdx.x;
  if (idx >= kB * kT) return;
  float za = out[o_z0 + (size_t)idx*3];
  float zb = out[o_z0 + (size_t)idx*3 + 1];
  float zc = out[o_z0 + (size_t)idx*3 + 2];
  float sld = 0.f;
  for (int f = 0; f < kNF; ++f) {
    float u0 = fu[f*3], u1 = fu[f*3+1], u2 = fu[f*3+2];
    float w0 = fw[f*3], w1 = fw[f*3+1], w2 = fw[f*3+2];
    float bb = fb[f];
    float wu = w0*u0 + w1*u1 + w2*u2;
    float coef = (-1.f + softplusf(wu) - wu) / (w0*w0 + w1*w1 + w2*w2 + 1e-8f);
    float uh0 = u0 + coef*w0, uh1 = u1 + coef*w1, uh2 = u2 + coef*w2;
    float h = tanhf(za*w0 + zb*w1 + zc*w2 + bb);
    za += h * uh0; zb += h * uh1; zc += h * uh2;
    float wuh = w0*uh0 + w1*uh1 + w2*uh2;
    float det = 1.f + (1.f - h*h) * wuh;
    sld += logf(fabsf(det) + 1e-8f);
  }
  out[o_z + (size_t)idx*3]     = za;
  out[o_z + (size_t)idx*3 + 1] = zb;
  out[o_z + (size_t)idx*3 + 2] = zc;
  out[o_sld + idx] = sld;
}

// fused prior: ph = relu(z_prev @ phW^T + phb); mu_p/lv_p. One wave per (b,t).
__global__ void prior_k(const float* __restrict__ z, const float* __restrict__ phW,
                        const float* __restrict__ phb, const float* __restrict__ pmuW,
                        const float* __restrict__ pmub, const float* __restrict__ plvW,
                        const float* __restrict__ plvb, float* __restrict__ mup,
                        float* __restrict__ lvp) {
  int idx = blockIdx.x * 8 + (threadIdx.x >> 5);
  int lane = threadIdx.x & 31;
  int t = idx % kT;
  float z0v = 0.f, z1v = 0.f, z2v = 0.f;
  if (t > 0) {
    const float* zp = z + (size_t)(idx - 1) * kZ;
    z0v = zp[0]; z1v = zp[1]; z2v = zp[2];
  }
  float a0=0,a1=0,a2=0,a3=0,a4=0,a5=0;
  for (int d = lane; d < kD; d += 32) {
    float ph = fmaf(phW[d*3], z0v, fmaf(phW[d*3+1], z1v, fmaf(phW[d*3+2], z2v, phb[d])));
    ph = fmaxf(ph, 0.f);
    a0 = fmaf(ph, pmuW[d],        a0);
    a1 = fmaf(ph, pmuW[kD + d],   a1);
    a2 = fmaf(ph, pmuW[2*kD + d], a2);
    a3 = fmaf(ph, plvW[d],        a3);
    a4 = fmaf(ph, plvW[kD + d],   a4);
    a5 = fmaf(ph, plvW[2*kD + d], a5);
  }
  for (int o = 16; o; o >>= 1) {
    a0 += __shfl_xor(a0, o, 32); a1 += __shfl_xor(a1, o, 32);
    a2 += __shfl_xor(a2, o, 32); a3 += __shfl_xor(a3, o, 32);
    a4 += __shfl_xor(a4, o, 32); a5 += __shfl_xor(a5, o, 32);
  }
  if (lane == 0) {
    size_t o3 = (size_t)idx * 3;
    mup[o3]   = a0 + pmub[0];
    mup[o3+1] = a1 + pmub[1];
    mup[o3+2] = a2 + pmub[2];
    lvp[o3]   = fminf(fmaxf(a3 + plvb[0], -8.f), 8.f);
    lvp[o3+1] = fminf(fmaxf(a4 + plvb[1], -8.f), 8.f);
    lvp[o3+2] = fminf(fmaxf(a5 + plvb[2], -8.f), 8.f);
  }
}

// Gaussian recon log-prob + last-step score, one wave per (b,t)
__global__ void recon_k(const float* __restrict__ x, const float* __restrict__ xmu,
                        const float* __restrict__ xstd, float* __restrict__ rlp,
                        float* __restrict__ score) {
  int idx = blockIdx.x * 8 + (threadIdx.x >> 5);
  int lane = threadIdx.x & 31;
  const float* xr = x    + (size_t)idx * kC;
  const float* mr = xmu  + (size_t)idx * kC;
  const float* sr = xstd + (size_t)idx * kC;
  float s = 0.f;
  for (int c = lane; c < kC; c += 32) {
    float d = (xr[c] - mr[c]) / sr[c];
    s += -0.5f * d * d - logf(sr[c]) - 0.5f * kLOG2PI;
  }
  for (int o = 16; o; o >>= 1) s += __shfl_xor(s, o, 32);
  if (lane == 0) {
    rlp[idx] = s;
    if ((idx % kT) == kT - 1) score[idx / kT] = -s;
  }
}

// ---------------------------------------------------------------------------
extern "C" void kernel_launch(void* const* d_in, const int* in_sizes, int n_in,
                              void* d_out, int out_size, void* d_ws, size_t ws_size,
                              hipStream_t stream) {
  (void)in_sizes; (void)n_in; (void)out_size; (void)ws_size;
  const float* x     = (const float*)d_in[0];
  const float* noise = (const float*)d_in[1];
  const float* qWih  = (const float*)d_in[2];
  const float* qWhh  = (const float*)d_in[3];
  const float* qbih  = (const float*)d_in[4];
  const float* qbhh  = (const float*)d_in[5];
  const float* pWih  = (const float*)d_in[6];
  const float* pWhh  = (const float*)d_in[7];
  const float* pbih  = (const float*)d_in[8];
  const float* pbhh  = (const float*)d_in[9];
  const float* qhW   = (const float*)d_in[10];
  const float* qhb   = (const float*)d_in[11];
  const float* qmuW  = (const float*)d_in[12];
  const float* qmub  = (const float*)d_in[13];
  const float* qlvW  = (const float*)d_in[14];
  const float* qlvb  = (const float*)d_in[15];
  const float* phW   = (const float*)d_in[16];
  const float* phb   = (const float*)d_in[17];
  const float* pmuW  = (const float*)d_in[18];
  const float* pmub  = (const float*)d_in[19];
  const float* plvW  = (const float*)d_in[20];
  const float* plvb  = (const float*)d_in[21];
  const float* dhW   = (const float*)d_in[22];
  const float* dhb   = (const float*)d_in[23];
  const float* xmuW  = (const float*)d_in[24];
  const float* xmub  = (const float*)d_in[25];
  const float* xlsW  = (const float*)d_in[26];
  const float* xlsb  = (const float*)d_in[27];
  const float* fu    = (const float*)d_in[28];
  const float* fw    = (const float*)d_in[29];
  const float* fb    = (const float*)d_in[30];
  float* out = (float*)d_out;

  char* ws = (char*)d_ws;
  size_t off = 0;
  auto alloc = [&](size_t bytes) -> char* {
    char* p = ws + off;
    off += (bytes + 255) & ~size_t(255);
    return p;
  };

  _Float16* qWhh_h = (_Float16*)alloc((size_t)kG3p * kHp * 2);   // 1.54 MB
  _Float16* pWhh_h = (_Float16*)alloc((size_t)kG3p * kHp * 2);
  _Float16* qhW_h  = (_Float16*)alloc((size_t)kHp * kHp * 2);
  _Float16* dhW_h  = (_Float16*)alloc((size_t)kHp * kHp * 2);
  _Float16* xmuW_h = (_Float16*)alloc((size_t)48 * kHp * 2);
  _Float16* xlsW_h = (_Float16*)alloc((size_t)48 * kHp * 2);
  float*    G      = (float*)   alloc((size_t)kB * kG3p * 4);    // per-step gates
  float*    qh     = (float*)   alloc((size_t)kB * kHp * 4);
  // zero-initialized region (contiguous):
  char* zero_base = ws + off;
  float*    henc32 = (float*)   alloc((size_t)2 * kB * kHp * 4);
  _Float16* henc16 = (_Float16*)alloc((size_t)2 * kB * kHp * 2);
  float*    hdec32 = (float*)   alloc((size_t)2 * kB * kHp * 4);
  _Float16* hdec16 = (_Float16*)alloc((size_t)2 * kB * kHp * 2);
  float*    zprev  = (float*)   alloc((size_t)2 * kB * 4 * 4);
  size_t zero_bytes = (size_t)((ws + off) - zero_base);
  _Float16* hseq   = (_Float16*)alloc((size_t)kT * kB * kHp * 2); // reused as dec_seq
  _Float16* dh     = (_Float16*)alloc((size_t)kT * kB * kHp * 2);

  // 0) zero recurrent state + zprev
  {
    int n16 = (int)(zero_bytes / 16);
    zero_k<<<(n16 + 255) / 256, 256, 0, stream>>>((uint4*)zero_base, n16);
  }
  // 1) pack weights to f16 [N_pad, 512]
  pack_w_k<<<(kG3p * kHp + 255) / 256, 256, 0, stream>>>(qWhh, qWhh_h, kG3, kH, kHp, kG3p * kHp);
  pack_w_k<<<(kG3p * kHp + 255) / 256, 256, 0, stream>>>(pWhh, pWhh_h, kG3, kH, kHp, kG3p * kHp);
  pack_w_k<<<(kHp * kHp + 255) / 256, 256, 0, stream>>>(qhW, qhW_h, kD, 503, kHp, kHp * kHp);
  pack_w_k<<<(kHp * kHp + 255) / 256, 256, 0, stream>>>(dhW, dhW_h, kD, kH, kHp, kHp * kHp);
  pack_w_k<<<(48 * kHp + 255) / 256, 256, 0, stream>>>(xmuW, xmuW_h, kC, kH, kHp, 48 * kHp);
  pack_w_k<<<(48 * kHp + 255) / 256, 256, 0, stream>>>(xlsW, xlsW_h, kC, kH, kHp, 48 * kHp);

  const size_t hstep = (size_t)kB * kHp;  // 262144 elements per state buffer / per t

  // 2) encoder GRU
  for (int t = 0; t < kT; ++t) {
    int cur = t & 1, nxt = cur ^ 1;
    gemm_gates_k<<<188, 256, 0, stream>>>(henc16 + cur * hstep, qWhh_h, G);
    gru_update_enc_k<<<1024, 256, 0, stream>>>(x, t, qWih, qbih, qbhh, G,
                                               henc32 + cur * hstep, henc32 + nxt * hstep,
                                               henc16 + nxt * hstep, hseq + (size_t)t * hstep);
  }
  // 3) sequential posterior + sampling
  for (int t = 0; t < kT; ++t) {
    int cur = t & 1, nxt = cur ^ 1;
    gemm_qh_k<<<32, 256, 0, stream>>>(hseq + (size_t)t * hstep, qhW_h, qhW, qhb,
                                      zprev + cur * ((size_t)kB * 4), qh);
    post_head_k<<<64, 256, 0, stream>>>(qh, qmuW, qmub, qlvW, qlvb, noise, t,
                                        zprev + nxt * ((size_t)kB * 4), out);
  }
  // 4) planar flows
  flows_k<<<(kB * kT + 255) / 256, 256, 0, stream>>>(fu, fw, fb, out);
  // 5) decoder GRU over z (dec_seq reuses hseq buffer)
  for (int t = 0; t < kT; ++t) {
    int cur = t & 1, nxt = cur ^ 1;
    gemm_gates_k<<<188, 256, 0, stream>>>(hdec16 + cur * hstep, pWhh_h, G);
    gru_update_dec_k<<<1024, 256, 0, stream>>>(out + o_z, t, pWih, pbih, pbhh, G,
                                               hdec32 + cur * hstep, hdec32 + nxt * hstep,
                                               hdec16 + nxt * hstep, hseq + (size_t)t * hstep);
  }
  // 6) decoder head
  gemm_dh_k<<<3200, 256, 0, stream>>>(hseq, dhW_h, dhb, dh);
  gemm_head_k<<<400, 256, 0, stream>>>(dh, xmuW_h, xmub, out + o_xmu, 0);
  gemm_head_k<<<400, 256, 0, stream>>>(dh, xlsW_h, xlsb, out + o_xstd, 1);
  // 7) recon log-prob + score
  recon_k<<<6400, 256, 0, stream>>>(x, out + o_xmu, out + o_xstd, out + o_rlp, out + o_score);
  // 8) connected prior
  prior_k<<<6400, 256, 0, stream>>>(out + o_z, phW, phb, pmuW, pmub, plvW, plvb,
                                    out + o_mup, out + o_lvp);
}
